// VectorQuantizer_86887188398665
// MI455X (gfx1250) — compile-verified, matrix operand order checked by reading, not run on hardware
//
#include <hip/hip_runtime.h>
#include <hip/hip_bf16.h>
#include <stdint.h>

#define N_TOKENS        262144
#define DIM             64
#define NCODES          1024
#define CODES_PER_PASS  256
#define NPASS           (NCODES / CODES_PER_PASS)
#define BLOCKS_PER_PASS (CODES_PER_PASS / 16)
#define WAVES_PER_WG    8
#define THREADS         (WAVES_PER_WG * 32)
#define TOKENS_PER_WG   (WAVES_PER_WG * 16)
#define NUM_WG          (N_TOKENS / TOKENS_PER_WG)

#define Q_OFF    0
#define LOSS_OFF (N_TOKENS * DIM)
#define IDX_OFF  (LOSS_OFF + 1)

// Workspace layout (bytes): frag-ordered bf16 hi, bf16 lo, then f32 |c|^2.
#define WS_HI_ELT  (NCODES * DIM)                  // 65536 ushorts
#define WS_LO_OFFB (WS_HI_ELT * 2)                 // 131072 B
#define WS_C2_OFFB (2 * WS_HI_ELT * 2)             // 262144 B
#define WS_NEEDED  (WS_C2_OFFB + NCODES * 4)       // 266240 B

// 1 = use global_load_async_to_lds_b128 staging; 0 = plain b128 copy staging.
#define VQ_ASYNC 1

typedef __attribute__((ext_vector_type(16))) __bf16 v16bf;
typedef __attribute__((ext_vector_type(8)))  float  v8f;

union BF16Frag { v16bf v; __bf16 e[16]; };
union F32Acc   { v8f   v; float  f[8];  };

// ---------------------------------------------------------------------------
// Prep: split codebook into bf16 hi/lo in exact WMMA B-fragment order + |c|^2.
// One thread per codebook row; runs once per launch (4 workgroups).
// ---------------------------------------------------------------------------
__global__ __launch_bounds__(256) void vq_prep_kernel(
    const float* __restrict__ CB,
    unsigned short* __restrict__ wsHI,
    unsigned short* __restrict__ wsLO,
    float* __restrict__ wsC2)
{
    const int g    = blockIdx.x * 256 + threadIdx.x;  // codebook row 0..1023
    const int pass = g >> 8;
    const int r    = g & 255;
    const int b    = r >> 4;
    const int rl   = r & 15;
    const float* cr = CB + (size_t)g * DIM;

    float xv[DIM];
    float ss = 0.0f;
#pragma unroll
    for (int k = 0; k < DIM; k += 4) {
        float4 p = *(const float4*)(cr + k);
        xv[k] = p.x; xv[k + 1] = p.y; xv[k + 2] = p.z; xv[k + 3] = p.w;
        ss += p.x * p.x + p.y * p.y + p.z * p.z + p.w * p.w;
    }

#pragma unroll
    for (int c = 0; c < 2; ++c) {
#pragma unroll
        for (int h = 0; h < 2; ++h) {
            alignas(16) unsigned short hi[16], lo[16];
#pragma unroll
            for (int j = 0; j < 16; ++j) {
                float f = xv[c * 32 + h * 16 + j];
                __bf16 bh = (__bf16)f;
                __bf16 bl = (__bf16)(f - (float)bh);
                hi[j] = __builtin_bit_cast(unsigned short, bh);
                lo[j] = __builtin_bit_cast(unsigned short, bl);
            }
            const size_t idx = (size_t)pass * 16384 + (size_t)(b * 2 + c) * 512
                             + (size_t)(rl + 16 * h) * 16;
            *(uint4*)(wsHI + idx)     = *(const uint4*)&hi[0];
            *(uint4*)(wsHI + idx + 8) = *(const uint4*)&hi[8];
            *(uint4*)(wsLO + idx)     = *(const uint4*)&lo[0];
            *(uint4*)(wsLO + idx + 8) = *(const uint4*)&lo[8];
        }
    }
    wsC2[g] = ss;
}

// ---------------------------------------------------------------------------
// Main kernel.
// ---------------------------------------------------------------------------
__global__ __launch_bounds__(THREADS) void vq_wmma_kernel(
    const float* __restrict__ X,    // [N_TOKENS, 64]
    const float* __restrict__ CB,   // [1024, 64]
    float* __restrict__ out,        // [N*64 quantized][1 loss][N indices]
    const unsigned short* __restrict__ wsHI,
    const unsigned short* __restrict__ wsLO,
    const float* __restrict__ wsC2,
    int use_ws)
{
    __shared__ __align__(16) unsigned short sBH[BLOCKS_PER_PASS * 2 * 512];
    __shared__ __align__(16) unsigned short sBL[BLOCKS_PER_PASS * 2 * 512];
    __shared__ float sC2[CODES_PER_PASS];

    const int tid  = threadIdx.x;
    const int lane = tid & 31;
    const int wave = tid >> 5;
    const int base = (blockIdx.x * WAVES_PER_WG + wave) * 16;

    // ---- Persistent A fragments: x tile (16 tokens x 64), hi/lo bf16 split.
    BF16Frag ah[2], al[2];
    {
        const int row   = base + (lane & 15);
        const int khalf = (lane >> 4) * 16;
#pragma unroll
        for (int c = 0; c < 2; ++c) {
            const float* xr = X + (size_t)row * DIM + c * 32 + khalf;
            float xv[16];
#pragma unroll
            for (int j = 0; j < 16; j += 4) {
                float4 p = *(const float4*)(xr + j);
                xv[j] = p.x; xv[j + 1] = p.y; xv[j + 2] = p.z; xv[j + 3] = p.w;
            }
#pragma unroll
            for (int j = 0; j < 16; ++j) {
                __bf16 h = (__bf16)xv[j];
                ah[c].e[j] = h;
                al[c].e[j] = (__bf16)(xv[j] - (float)h);
            }
        }
    }

    float bestVal[8];
    int   bestIdx[8];
#pragma unroll
    for (int r = 0; r < 8; ++r) { bestVal[r] = 3.0e38f; bestIdx[r] = 0; }

    for (int pass = 0; pass < NPASS; ++pass) {
        if (use_ws) {
            // ---- Fast staging: pure copy of pre-split, pre-swizzled fragments.
            const unsigned short* gH = wsHI + (size_t)pass * (CODES_PER_PASS * DIM);
            const unsigned short* gL = wsLO + (size_t)pass * (CODES_PER_PASS * DIM);
#if VQ_ASYNC
            const unsigned ldsH = (unsigned)(uintptr_t)&sBH[0];
            const unsigned ldsL = (unsigned)(uintptr_t)&sBL[0];
#pragma unroll
            for (int i = 0; i < 8; ++i) {
                const unsigned bo = (unsigned)((i * THREADS + tid) * 16);
                const unsigned dH = ldsH + bo;
                const unsigned dL = ldsL + bo;
                asm volatile("global_load_async_to_lds_b128 %0, %1, %2"
                             :: "v"(dH), "v"(bo), "s"(gH) : "memory");
                asm volatile("global_load_async_to_lds_b128 %0, %1, %2"
                             :: "v"(dL), "v"(bo), "s"(gL) : "memory");
            }
            sC2[tid] = wsC2[pass * CODES_PER_PASS + tid];
            asm volatile("s_wait_asynccnt 0" ::: "memory");
#else
#pragma unroll
            for (int i = 0; i < 8; ++i) {
                const int eo = (i * THREADS + tid) * 8;   // ushort index, 16B steps
                *(uint4*)&sBH[eo] = *(const uint4*)(gH + eo);
                *(uint4*)&sBL[eo] = *(const uint4*)(gL + eo);
            }
            sC2[tid] = wsC2[pass * CODES_PER_PASS + tid];
#endif
        } else {
            // ---- Fallback staging: split + swizzle in-kernel.
            const int r  = tid;
            const int g  = pass * CODES_PER_PASS + r;
            const float* cr = CB + (size_t)g * DIM;
            const int b  = r >> 4;
            const int rl = r & 15;
            float ss = 0.0f;
#pragma unroll
            for (int k0 = 0; k0 < DIM; k0 += 4) {
                float4 p = *(const float4*)(cr + k0);
                float fv[4] = {p.x, p.y, p.z, p.w};
#pragma unroll
                for (int j = 0; j < 4; ++j) {
                    const int k = k0 + j;
                    float f = fv[j];
                    ss += f * f;
                    __bf16 h = (__bf16)f;
                    __bf16 l = (__bf16)(f - (float)h);
                    const int c   = k >> 5;
                    const int li  = rl + ((k >> 4) & 1) * 16;
                    const int pos = ((b * 2 + c) * 512) + li * 16 + (k & 15);
                    sBH[pos] = __builtin_bit_cast(unsigned short, h);
                    sBL[pos] = __builtin_bit_cast(unsigned short, l);
                }
            }
            sC2[r] = ss;
        }
        __syncthreads();

        // ---- WMMA scores: 16 blocks of 16 codes, 6 chained bf16 WMMAs each.
#pragma unroll 2
        for (int blk = 0; blk < BLOCKS_PER_PASS; ++blk) {
            const int fbase = blk * 2 * 512 + lane * 16;
            BF16Frag bh0, bh1, bl0, bl1;
            bh0.v = *(const v16bf*)&sBH[fbase];
            bh1.v = *(const v16bf*)&sBH[fbase + 512];
            bl0.v = *(const v16bf*)&sBL[fbase];
            bl1.v = *(const v16bf*)&sBL[fbase + 512];
            const float c2v = sC2[blk * 16 + (lane & 15)];

            v8f accv = {};
            accv = __builtin_amdgcn_wmma_f32_16x16x32_bf16(false, ah[0].v, false, bl0.v, (short)0, accv, false, false);
            accv = __builtin_amdgcn_wmma_f32_16x16x32_bf16(false, ah[1].v, false, bl1.v, (short)0, accv, false, false);
            accv = __builtin_amdgcn_wmma_f32_16x16x32_bf16(false, al[0].v, false, bh0.v, (short)0, accv, false, false);
            accv = __builtin_amdgcn_wmma_f32_16x16x32_bf16(false, al[1].v, false, bh1.v, (short)0, accv, false, false);
            accv = __builtin_amdgcn_wmma_f32_16x16x32_bf16(false, ah[0].v, false, bh0.v, (short)0, accv, false, false);
            accv = __builtin_amdgcn_wmma_f32_16x16x32_bf16(false, ah[1].v, false, bh1.v, (short)0, accv, false, false);
            F32Acc acc; acc.v = accv;

            const int n = pass * CODES_PER_PASS + blk * 16 + (lane & 15);
#pragma unroll
            for (int r = 0; r < 8; ++r) {
                float s = c2v - 2.0f * acc.f[r];
                if (s < bestVal[r]) { bestVal[r] = s; bestIdx[r] = n; }
            }
        }
        __syncthreads();
    }

    // ---- Argmin reduction over the 16 lanes holding each row.
#pragma unroll
    for (int r = 0; r < 8; ++r) {
#pragma unroll
        for (int mask = 1; mask <= 8; mask <<= 1) {
            float ov = __shfl_xor(bestVal[r], mask, 32);
            int   oi = __shfl_xor(bestIdx[r], mask, 32);
            if (ov < bestVal[r] || (ov == bestVal[r] && oi < bestIdx[r])) {
                bestVal[r] = ov; bestIdx[r] = oi;
            }
        }
    }

    // ---- Gather winning rows, write quantized + indices, accumulate loss.
    float sumsq = 0.0f;
    const int e = lane * 2;
#pragma unroll
    for (int m = 0; m < 8; ++m) {
        const int iLo = __shfl(bestIdx[m], 0, 32);
        const int iHi = __shfl(bestIdx[m], 16, 32);
        const int tLo = base + m;
        const int tHi = base + m + 8;

        float2 q0 = *(const float2*)(CB + (size_t)iLo * DIM + e);
        float2 x0 = *(const float2*)(X  + (size_t)tLo * DIM + e);
        *(float2*)(out + Q_OFF + (size_t)tLo * DIM + e) = q0;
        float d0 = q0.x - x0.x, d1 = q0.y - x0.y;
        sumsq += d0 * d0 + d1 * d1;

        float2 q1 = *(const float2*)(CB + (size_t)iHi * DIM + e);
        float2 x1 = *(const float2*)(X  + (size_t)tHi * DIM + e);
        *(float2*)(out + Q_OFF + (size_t)tHi * DIM + e) = q1;
        float d2 = q1.x - x1.x, d3 = q1.y - x1.y;
        sumsq += d2 * d2 + d3 * d3;

        if (lane == 0) {
            out[IDX_OFF + tLo] = (float)iLo;
            out[IDX_OFF + tHi] = (float)iHi;
        }
    }
#pragma unroll
    for (int mask = 1; mask < 32; mask <<= 1)
        sumsq += __shfl_xor(sumsq, mask, 32);
    if (lane == 0)
        atomicAdd(out + LOSS_OFF, sumsq * (1.25f / (float)(N_TOKENS * DIM)));
}

extern "C" void kernel_launch(void* const* d_in, const int* in_sizes, int n_in,
                              void* d_out, int out_size, void* d_ws, size_t ws_size,
                              hipStream_t stream) {
    const float* X  = (const float*)d_in[0];
    const float* CB = (const float*)d_in[1];
    float* out = (float*)d_out;
    (void)in_sizes; (void)n_in; (void)out_size;

    const int use_ws = (d_ws != nullptr && ws_size >= (size_t)WS_NEEDED) ? 1 : 0;
    unsigned short* wsHI = (unsigned short*)d_ws;
    unsigned short* wsLO = (unsigned short*)((char*)d_ws + WS_LO_OFFB);
    float*          wsC2 = (float*)((char*)d_ws + WS_C2_OFFB);

    // Loss accumulator must start at zero every call (atomics add into it).
    hipMemsetAsync((char*)d_out + (size_t)LOSS_OFF * sizeof(float), 0, sizeof(float), stream);

    if (use_ws)
        vq_prep_kernel<<<NCODES / 256, 256, 0, stream>>>(CB, wsHI, wsLO, wsC2);

    vq_wmma_kernel<<<NUM_WG, THREADS, 0, stream>>>(X, CB, out, wsHI, wsLO, wsC2, use_ws);
}
